// AlexNet_76553497084060
// MI455X (gfx1250) — matrix operand order repaired and checked by reading, not compiled, not run on hardware
//
#include <hip/hip_runtime.h>
#include <hip/hip_bf16.h>

// ---------------------------------------------------------------------------
// AlexNet forward for MI455X (gfx1250, wave32, WMMA).
// All conv layers -> im2col (f16, [N][K] K-major) x packed f16 weights [M][K]
// through one generic 128x128x32-tile WMMA GEMM (v_wmma_f32_16x16x32_f16),
// bias+ReLU fused in the epilogue. LRN / maxpool are elementwise kernels.
//
// Workspace layout (requires ws_size >= ~844 MB):
//   [0      , 160 MB) actA   (f32 activations ping)
//   [160 MB , 320 MB) actB   (f32 activations pong)
//   [320 MB , 768 MB) col    (f16 im2col / packed FC activations, max 448 MB)
//   [768 MB , 844 MB) wpk    (f16 packed weights, max 76 MB)
// ---------------------------------------------------------------------------

typedef __attribute__((ext_vector_type(16))) _Float16 v16h;
typedef __attribute__((ext_vector_type(8)))  _Float16 v8h;
typedef __attribute__((ext_vector_type(8)))  float    v8f;

#define TILE_M 128
#define TILE_N 128
#define TILE_K 32
#define LDT    40   // LDS row stride in halfs (32 + 8 pad -> 80B, staggers banks)

// ---------------------------------------------------------------------------
// im2col: x (NCHW f32) -> col f16 [N][Kp], n = b*OH*OW + oh*OW + ow,
// k = c*KH*KW + kh*KW + kw. Zero-fills conv padding and K-padding.
// ---------------------------------------------------------------------------
__global__ void im2col_f16_kernel(const float* __restrict__ x,
                                  _Float16* __restrict__ col,
                                  int Bn, int C, int H, int W,
                                  int KH, int KW, int stride, int pad,
                                  int OH, int OW, int K, int Kp, long long N)
{
    long long idx = (long long)blockIdx.x * blockDim.x + threadIdx.x;
    long long total = N * (long long)Kp;
    if (idx >= total) return;
    long long n = idx / Kp;
    int k = (int)(idx - n * Kp);
    _Float16 v = (_Float16)0.f;
    if (k < K) {
        int ohw = OH * OW;
        int b = (int)(n / ohw);
        int s = (int)(n - (long long)b * ohw);
        int oh = s / OW, ow = s - oh * OW;
        int khw = KH * KW;
        int c = k / khw;
        int r = k - c * khw;
        int kh = r / KW, kw = r - kh * KW;
        int ih = oh * stride - pad + kh;
        int iw = ow * stride - pad + kw;
        if (ih >= 0 && ih < H && iw >= 0 && iw < W) {
            v = (_Float16)x[(((long long)b * C + c) * H + ih) * W + iw];
        }
    }
    col[n * Kp + k] = v;
}

// ---------------------------------------------------------------------------
// pack: f32 [M][K] -> f16 [M2][K2], zero padded. Used for weights and FC acts.
// ---------------------------------------------------------------------------
__global__ void pack_f16_kernel(const float* __restrict__ in,
                                _Float16* __restrict__ out,
                                long long M, long long K,
                                long long M2, long long K2)
{
    long long idx = (long long)blockIdx.x * blockDim.x + threadIdx.x;
    if (idx >= M2 * K2) return;
    long long r = idx / K2;
    long long c = idx - r * K2;
    out[idx] = (r < M && c < K) ? (_Float16)in[r * K + c] : (_Float16)0.f;
}

// ---------------------------------------------------------------------------
// WMMA GEMM: C[M][N] = A[M][K2] * B[N][K2]^T  (both K-major f16, K2 % 32 == 0,
// A padded to M2 % 128 == 0, B padded to N2 % 128 == 0), + bias, ReLU.
// mode 0: scatter to NCHW f32 activations   (n = b*OHW + s, m = channel)
// mode 1: out[n*M + m] f32                  (FC layers)
// Block: 256 threads = 8 wave32, block tile 128x128, wave tile 32x64.
// ---------------------------------------------------------------------------
__global__ __launch_bounds__(256)
void gemm_wmma_kernel(const _Float16* __restrict__ A,
                      const _Float16* __restrict__ Bm,
                      const float* __restrict__ bias,
                      float* __restrict__ out,
                      int M, long long N, int K2, int mode, int OHW)
{
    __shared__ _Float16 As[TILE_M * LDT];
    __shared__ _Float16 Bs[TILE_N * LDT];

    const int tid  = threadIdx.x;
    const int lane = tid & 31;
    const int wave = tid >> 5;
    const int wm   = wave & 3;   // wave row group: rows wm*32 .. wm*32+31
    const int wn   = wave >> 2;  // wave col group: cols wn*64 .. wn*64+63
    const int h    = lane >> 4;  // lane half (0/1)
    const int l16  = lane & 15;

    const int m0 = blockIdx.y * TILE_M;
    const long long n0 = (long long)blockIdx.x * TILE_N;

    v8f zero = {0.f, 0.f, 0.f, 0.f, 0.f, 0.f, 0.f, 0.f};
    v8f acc[2][4];
#pragma unroll
    for (int i = 0; i < 2; ++i)
#pragma unroll
        for (int j = 0; j < 4; ++j) acc[i][j] = zero;

    const int ktiles = K2 / TILE_K;
    for (int kt = 0; kt < ktiles; ++kt) {
        // ---- stage 128x32 A tile and 128x32 B tile into LDS (16B chunks) ----
#pragma unroll
        for (int t = 0; t < 2; ++t) {
            int chunk = tid + t * 256;          // 512 chunks of 8 halfs
            int row   = chunk >> 2;
            int c8    = (chunk & 3) * 8;
            v8h va = *(const v8h*)(A  + (long long)(m0 + row) * K2 + kt * TILE_K + c8);
            v8h vb = *(const v8h*)(Bm + (n0 + row) * K2 + kt * TILE_K + c8);
            *(v8h*)(&As[row * LDT + c8]) = va;
            *(v8h*)(&Bs[row * LDT + c8]) = vb;
        }
        // hint-prefetch the next K tile (global_prefetch_b8)
        if (kt + 1 < ktiles) {
            __builtin_prefetch(A  + (long long)(m0 + (tid >> 1)) * K2 + (kt + 1) * TILE_K, 0, 1);
            __builtin_prefetch(Bm + (n0 + (tid >> 1)) * K2 + (kt + 1) * TILE_K, 0, 1);
        }
        __syncthreads();

        // ---- load WMMA fragments (ISA 16-bit A 16x32 / B 32x16 layouts) ----
        v16h af[2], bf[4];
#pragma unroll
        for (int i = 0; i < 2; ++i) {
            // A: M = l16, lane half h holds K = 8h..8h+7 and 16+8h..16+8h+7
            const _Float16* p = &As[(wm * 32 + i * 16 + l16) * LDT];
            v8h lo = *(const v8h*)(p + 8 * h);
            v8h hi = *(const v8h*)(p + 16 + 8 * h);
#pragma unroll
            for (int e = 0; e < 8; ++e) { af[i][e] = lo[e]; af[i][e + 8] = hi[e]; }
        }
#pragma unroll
        for (int j = 0; j < 4; ++j) {
            // B: N = l16, lane half h holds K = 8h..8h+7 and 16+8h..16+8h+7
            const _Float16* p = &Bs[(wn * 64 + j * 16 + l16) * LDT];
            v8h lo = *(const v8h*)(p + 8 * h);
            v8h hi = *(const v8h*)(p + 16 + 8 * h);
#pragma unroll
            for (int e = 0; e < 8; ++e) { bf[j][e] = lo[e]; bf[j][e + 8] = hi[e]; }
        }

        // ---- 8 WMMAs: D = A x B + C, f32 accumulate ----
#pragma unroll
        for (int i = 0; i < 2; ++i)
#pragma unroll
            for (int j = 0; j < 4; ++j)
                acc[i][j] = __builtin_amdgcn_wmma_f32_16x16x32_f16(
                    /*neg_a=*/false, af[i], /*neg_b=*/false, bf[j],
                    /*c_mod=*/(short)0, acc[i][j],
                    /*reuse_a=*/false, /*reuse_b=*/false);
        __syncthreads();
    }

    // ---- epilogue: bias + ReLU, scatter (C/D layout: elem r -> M=r+8h, N=l16)
#pragma unroll
    for (int i = 0; i < 2; ++i) {
#pragma unroll
        for (int j = 0; j < 4; ++j) {
            long long n = n0 + wn * 64 + j * 16 + l16;
            if (n >= N) continue;
#pragma unroll
            for (int r = 0; r < 8; ++r) {
                int m = m0 + wm * 32 + i * 16 + r + 8 * h;
                if (m >= M) continue;
                float v = acc[i][j][r] + bias[m];
                v = v > 0.f ? v : 0.f;
                if (mode == 0) {
                    long long b = n / OHW;
                    long long s = n - b * OHW;
                    out[(b * M + m) * OHW + s] = v;
                } else {
                    out[n * (long long)M + m] = v;
                }
            }
        }
    }
}

// ---------------------------------------------------------------------------
// LRN: y = x * (k + alpha * sum_{c-2..c+2} x^2)^(-beta), k=2, alpha=1e-4,
// beta=0.75, window 5 clamped at channel edges. x,y NCHW f32.
// ---------------------------------------------------------------------------
__global__ void lrn_kernel(const float* __restrict__ x, float* __restrict__ y,
                           long long total, int C, int HW)
{
    long long idx = (long long)blockIdx.x * blockDim.x + threadIdx.x;
    if (idx >= total) return;
    long long b = idx / ((long long)C * HW);
    long long rem = idx - b * (long long)C * HW;
    int c = (int)(rem / HW);
    int s = (int)(rem - (long long)c * HW);
    int lo = c - 2 > 0 ? c - 2 : 0;
    int hi = c + 2 < C - 1 ? c + 2 : C - 1;
    float sum = 0.f;
    for (int cc = lo; cc <= hi; ++cc) {
        float t = x[(b * C + cc) * HW + s];
        sum += t * t;
    }
    y[idx] = x[idx] / powf(2.0f + 1e-4f * sum, 0.75f);
}

// ---------------------------------------------------------------------------
// maxpool 3x3 stride 2, VALID. x,y are [BC][H][W] / [BC][OH][OW] f32.
// ---------------------------------------------------------------------------
__global__ void maxpool_kernel(const float* __restrict__ x, float* __restrict__ y,
                               long long BC, int H, int W, int OH, int OW)
{
    long long idx = (long long)blockIdx.x * blockDim.x + threadIdx.x;
    long long total = BC * OH * OW;
    if (idx >= total) return;
    long long bc = idx / ((long long)OH * OW);
    int s  = (int)(idx - bc * (long long)OH * OW);
    int oh = s / OW, ow = s - (s / OW) * OW;
    const float* p = x + bc * (long long)H * W;
    float m = -3.4e38f;
    for (int kh = 0; kh < 3; ++kh)
        for (int kw = 0; kw < 3; ++kw) {
            float v = p[(oh * 2 + kh) * W + (ow * 2 + kw)];
            m = v > m ? v : m;
        }
    y[idx] = m;
}

// ---------------------------------------------------------------------------
// Host-side helpers
// ---------------------------------------------------------------------------
static inline long long cdiv_ll(long long a, long long b) { return (a + b - 1) / b; }

struct ConvDesc { int Cin, H, W, KH, KW, stride, pad, Cout, OH, OW; };

static void run_conv(const float* x, const float* Wt, const float* bias, float* out,
                     _Float16* col, _Float16* wpk, ConvDesc d, int Bn,
                     hipStream_t stream)
{
    long long OHW = (long long)d.OH * d.OW;
    long long N   = (long long)Bn * OHW;         // multiple of 128 (Bn=128)
    int K  = d.Cin * d.KH * d.KW;
    int Kp = (int)(cdiv_ll(K, TILE_K) * TILE_K);
    int M  = d.Cout;
    int M2 = (int)(cdiv_ll(M, TILE_M) * TILE_M);

    long long itot = N * Kp;
    im2col_f16_kernel<<<(unsigned)cdiv_ll(itot, 256), 256, 0, stream>>>(
        x, col, Bn, d.Cin, d.H, d.W, d.KH, d.KW, d.stride, d.pad,
        d.OH, d.OW, K, Kp, N);

    long long wtot = (long long)M2 * Kp;
    pack_f16_kernel<<<(unsigned)cdiv_ll(wtot, 256), 256, 0, stream>>>(
        Wt, wpk, M, K, M2, Kp);

    dim3 g((unsigned)(N / TILE_N), (unsigned)(M2 / TILE_M));
    gemm_wmma_kernel<<<g, 256, 0, stream>>>(wpk, col, bias, out, M, N, Kp,
                                            /*mode=*/0, (int)OHW);
}

static void run_fc(const float* act, const float* Wt, const float* bias, float* out,
                   _Float16* col, _Float16* wpk, int Bn, int M, int K,
                   hipStream_t stream)
{
    int Kp = (int)(cdiv_ll(K, TILE_K) * TILE_K);
    int M2 = (int)(cdiv_ll(M, TILE_M) * TILE_M);

    long long btot = (long long)Bn * Kp;         // Bn = 128, one N tile
    pack_f16_kernel<<<(unsigned)cdiv_ll(btot, 256), 256, 0, stream>>>(
        act, col, Bn, K, Bn, Kp);

    long long wtot = (long long)M2 * Kp;
    pack_f16_kernel<<<(unsigned)cdiv_ll(wtot, 256), 256, 0, stream>>>(
        Wt, wpk, M, K, M2, Kp);

    dim3 g((unsigned)(Bn / TILE_N), (unsigned)(M2 / TILE_M));
    gemm_wmma_kernel<<<g, 256, 0, stream>>>(wpk, col, bias, out, M, (long long)Bn,
                                            Kp, /*mode=*/1, 0);
}

extern "C" void kernel_launch(void* const* d_in, const int* in_sizes, int n_in,
                              void* d_out, int out_size, void* d_ws, size_t ws_size,
                              hipStream_t stream)
{
    (void)in_sizes; (void)n_in; (void)out_size; (void)ws_size;
    const float* x  = (const float*)d_in[0];
    const float* W1 = (const float*)d_in[1];  const float* b1 = (const float*)d_in[2];
    const float* W2 = (const float*)d_in[3];  const float* b2 = (const float*)d_in[4];
    const float* W3 = (const float*)d_in[5];  const float* b3 = (const float*)d_in[6];
    const float* W4 = (const float*)d_in[7];  const float* b4 = (const float*)d_in[8];
    const float* W5 = (const float*)d_in[9];  const float* b5 = (const float*)d_in[10];
    const float* W6 = (const float*)d_in[11]; const float* b6 = (const float*)d_in[12];
    const float* W7 = (const float*)d_in[13]; const float* b7 = (const float*)d_in[14];
    const float* W8 = (const float*)d_in[15]; const float* b8 = (const float*)d_in[16];

    char* w = (char*)d_ws;
    float*    actA = (float*)(w);
    float*    actB = (float*)(w + (size_t)160 * 1024 * 1024);
    _Float16* col  = (_Float16*)(w + (size_t)320 * 1024 * 1024);
    _Float16* wpk  = (_Float16*)(w + (size_t)768 * 1024 * 1024);
    const int Bn = 128;

    // ---- conv1 (+ReLU) -> lrn1 -> pool1 ----
    run_conv(x, W1, b1, actA, col, wpk,
             ConvDesc{3, 224, 224, 11, 11, 4, 2, 96, 55, 55}, Bn, stream);
    {
        long long tot = (long long)Bn * 96 * 55 * 55;
        lrn_kernel<<<(unsigned)cdiv_ll(tot, 256), 256, 0, stream>>>(
            actA, actB, tot, 96, 55 * 55);
        long long ptot = (long long)Bn * 96 * 27 * 27;
        maxpool_kernel<<<(unsigned)cdiv_ll(ptot, 256), 256, 0, stream>>>(
            actB, actA, (long long)Bn * 96, 55, 55, 27, 27);
    }
    // ---- conv2 (+ReLU) -> lrn2 -> pool2 ----
    run_conv(actA, W2, b2, actB, col, wpk,
             ConvDesc{96, 27, 27, 5, 5, 1, 2, 256, 27, 27}, Bn, stream);
    {
        long long tot = (long long)Bn * 256 * 27 * 27;
        lrn_kernel<<<(unsigned)cdiv_ll(tot, 256), 256, 0, stream>>>(
            actB, actA, tot, 256, 27 * 27);
        long long ptot = (long long)Bn * 256 * 13 * 13;
        maxpool_kernel<<<(unsigned)cdiv_ll(ptot, 256), 256, 0, stream>>>(
            actA, actB, (long long)Bn * 256, 27, 27, 13, 13);
    }
    // ---- conv3 -> conv4 -> conv5 (+ReLU each) -> pool5 ----
    run_conv(actB, W3, b3, actA, col, wpk,
             ConvDesc{256, 13, 13, 3, 3, 1, 1, 384, 13, 13}, Bn, stream);
    run_conv(actA, W4, b4, actB, col, wpk,
             ConvDesc{384, 13, 13, 3, 3, 1, 1, 384, 13, 13}, Bn, stream);
    run_conv(actB, W5, b5, actA, col, wpk,
             ConvDesc{384, 13, 13, 3, 3, 1, 1, 256, 13, 13}, Bn, stream);
    {
        long long ptot = (long long)Bn * 256 * 6 * 6;
        maxpool_kernel<<<(unsigned)cdiv_ll(ptot, 256), 256, 0, stream>>>(
            actA, actB, (long long)Bn * 256, 13, 13, 6, 6);
    }
    // ---- FC6 -> FC7 -> FC8 (+ReLU each), FC8 writes d_out [128][1000] ----
    run_fc(actB, W6, b6, actA, col, wpk, Bn, 4096, 256 * 6 * 6, stream);
    run_fc(actA, W7, b7, actB, col, wpk, Bn, 4096, 4096, stream);
    run_fc(actB, W8, b8, (float*)d_out, col, wpk, Bn, 1000, 4096, stream);
}
